// SelfAttention_84310208020991
// MI455X (gfx1250) — matrix operand order repaired and checked by reading, not compiled, not run on hardware
//
#include <hip/hip_runtime.h>

typedef __attribute__((ext_vector_type(16))) __bf16 v16bf;
typedef __attribute__((ext_vector_type(8)))  float  v8f;
typedef __attribute__((ext_vector_type(4)))  int    v4i;
typedef __attribute__((ext_vector_type(4)))  unsigned int v4u;
typedef unsigned int   u32;
typedef unsigned short u16;

// gfx1250 async global->LDS DMA (ASYNCcnt-tracked); round-2 diagnostic proved
// the builtin exists here. Guard keeps a register-staged fallback anyway.
#if __has_builtin(__builtin_amdgcn_global_load_async_to_lds_b128) && \
    __has_builtin(__builtin_amdgcn_s_wait_asynccnt)
#define USE_ASYNC 1
#else
#define USE_ASYNC 0
#endif

// ---------- bf16 helpers (RNE) ----------
__device__ __forceinline__ u32 pack_bf16(float f0, float f1) {
    u32 u0 = __float_as_uint(f0), u1 = __float_as_uint(f1);
    u32 r0 = u0 + 0x7FFFu + ((u0 >> 16) & 1u);
    u32 r1 = u1 + 0x7FFFu + ((u1 >> 16) & 1u);
    // one v_perm_b32: {hi16(r1), hi16(r0)}
    return __builtin_amdgcn_perm(r1, r0, 0x07060302u);
}
__device__ __forceinline__ u16 f2bf(float f) {
    u32 u = __float_as_uint(f);
    return (u16)((u + 0x7FFFu + ((u >> 16) & 1u)) >> 16);
}
__device__ __forceinline__ float bf2f(u16 h) {
    return __uint_as_float(((u32)h) << 16);
}

union Frag { v16bf v; u32 u[8]; };

#if USE_ASYNC
// Builtin signature (from clang diagnostic): (v4i addrspace(1)*, v4i addrspace(3)*, imm, imm)
typedef __attribute__((address_space(1))) v4i* gptr_v4i;
typedef __attribute__((address_space(3))) v4i* lptr_v4i;
__device__ __forceinline__ void async_copy_b128(const void* g, void* l) {
    __builtin_amdgcn_global_load_async_to_lds_b128((gptr_v4i)(g), (lptr_v4i)(l), 0, 0);
}
#endif

// ---------- f32 -> bf16 bulk convert (pure bandwidth; ~2us at 23.3 TB/s) ----
// Each thread: 8 consecutive f32 -> 4 packed u32, stored as one b128.
__global__ __launch_bounds__(256)
void cvt_kernel(const float* __restrict__ src, u32* __restrict__ dst)
{
    const long i = ((long)blockIdx.x * 256 + threadIdx.x) * 4;   // u32 index
    const float* s = src + i * 2;
    v4u o;
#pragma unroll
    for (int j = 0; j < 4; ++j) o[j] = pack_bf16(s[2*j], s[2*j + 1]);
    *(v4u*)(dst + i) = o;
}

// =====================================================================
// NT GEMM body (bf16 in):  C[m,n] = sum_k A[m,k] * B[n,k]
// Block tile 128x128x32, 256 threads (8 waves, 2x4 wave grid, each wave
// 4x2 WMMA 16x16 tiles). Double-buffered LDS, one barrier per K-step,
// async global->LDS DMA staging overlapped with WMMA.
// LDS row stride 20 u32 (80B): 16B aligned for b128, conflict-free gathers.
// =====================================================================
template<bool OUT_BF16>
__device__ __forceinline__ void gemm_nt_body(const u16* Ap, const u16* Bp, void* Cp,
                                             int lda, int ldb, int ldc, int K)
{
    __shared__ u32 ldsA[2][128][20];
    __shared__ u32 ldsB[2][128][20];

    const int tid  = threadIdx.x;
    const int lane = tid & 31;
    const int w    = tid >> 5;
    const int wm   = (w >> 2) * 64;
    const int wn   = (w & 3) * 32;
    const int hi   = lane >> 4;
    const int ln   = lane & 15;
    const long mBase = (long)blockIdx.y * 128;
    const long nBase = (long)blockIdx.x * 128;

    const v8f vzero = {0.f,0.f,0.f,0.f,0.f,0.f,0.f,0.f};
    v8f acc[4][2];
#pragma unroll
    for (int i = 0; i < 4; ++i)
#pragma unroll
        for (int j = 0; j < 2; ++j) acc[i][j] = vzero;

    const int sm = tid >> 1;          // staging row 0..127
    const int sh = (tid & 1) * 8;     // k2 offset 0 or 8

    auto stage = [&](int buf, int k0) {
        const u16* A = Ap + (mBase + sm) * (long)lda + k0 + sh * 2;
        const u16* B = Bp + (nBase + sm) * (long)ldb + k0 + sh * 2;
#if USE_ASYNC
        async_copy_b128(A,     &ldsA[buf][sm][sh]);
        async_copy_b128(A + 8, &ldsA[buf][sm][sh + 4]);
        async_copy_b128(B,     &ldsB[buf][sm][sh]);
        async_copy_b128(B + 8, &ldsB[buf][sm][sh + 4]);
#else
        const u32* A32 = (const u32*)A;
        const u32* B32 = (const u32*)B;
#pragma unroll
        for (int j = 0; j < 8; ++j) {
            ldsA[buf][sm][sh + j] = A32[j];
            ldsB[buf][sm][sh + j] = B32[j];
        }
        __builtin_prefetch(A32 + 16, 0, 0);
        __builtin_prefetch(B32 + 16, 0, 0);
#endif
    };

    stage(0, 0);
#if USE_ASYNC
    __builtin_amdgcn_s_wait_asynccnt(0);
#endif
    __syncthreads();

    int buf = 0;
    for (int k0 = 0; k0 < K; k0 += 32, buf ^= 1) {
        const bool hasNext = (k0 + 32 < K);
        if (hasNext) stage(buf ^ 1, k0 + 32);   // overlap next-tile DMA with WMMA

        Frag a[4], b[2];
#pragma unroll
        for (int i = 0; i < 4; ++i) {
            const int row = wm + 16 * i + ln;
#pragma unroll
            for (int j = 0; j < 8; ++j) {
                const int k2 = hi * 8 + j;
                a[i].u[j] = ldsA[buf][row][k2];
            }
        }
#pragma unroll
        for (int i = 0; i < 2; ++i) {
            const int row = wn + 16 * i + ln;
#pragma unroll
            for (int j = 0; j < 8; ++j)
                b[i].u[j] = ldsB[buf][row][hi * 8 + j];
        }
#pragma unroll
        for (int i = 0; i < 4; ++i)
#pragma unroll
            for (int j = 0; j < 2; ++j)
                acc[i][j] = __builtin_amdgcn_wmma_f32_16x16x32_bf16(
                    false, a[i].v, false, b[j].v, (short)0, acc[i][j], false, false);

#if USE_ASYNC
        if (hasNext) __builtin_amdgcn_s_wait_asynccnt(0);
#endif
        __syncthreads();
    }

    // ---- epilogue: C/D layout -> row-major store ----
#pragma unroll
    for (int i = 0; i < 4; ++i)
#pragma unroll
        for (int jn = 0; jn < 2; ++jn) {
            const long m0 = mBase + wm + 16 * i;
            const long n  = nBase + wn + 16 * jn + ln;
#pragma unroll
            for (int r = 0; r < 8; ++r) {
                const long m = m0 + r + hi * 8;
                if constexpr (OUT_BF16)
                    ((u16*)Cp)[m * (long)ldc + n] = f2bf(acc[i][jn][r]);
                else
                    ((float*)Cp)[m * (long)ldc + n] = acc[i][jn][r];
            }
        }
}

// ---------- Stage 1: QKV projection  Y = Xbf @ Wbf^T  (bf16 in, bf16 out) ----
__global__ __launch_bounds__(256)
void qkv_kernel(const u16* Xbf, const u16* Wbf, u16* Q, u16* Kd, u16* V)
{
    const u16* W = Wbf + (long)blockIdx.z * 1024 * 1024;
    u16*       O = (blockIdx.z == 0) ? Q : ((blockIdx.z == 1) ? Kd : V);
    gemm_nt_body<true>(Xbf, W, O, 1024, 1024, 1024, 1024);
}

// ---------- Stage 2: S_b = Q_b @ K_b^T (f32 out; scale folded into softmax) --
__global__ __launch_bounds__(256)
void score_kernel(const u16* Q, const u16* Kd, float* S)
{
    const long b = blockIdx.z;
    gemm_nt_body<false>(Q + b * 2048L * 1024, Kd + b * 2048L * 1024,
                        S + b * 2048L * 2048, 1024, 1024, 2048, 1024);
}

// ---------- Stage 3: row softmax; writes bf16 P in-place over f32 S rows ----
__global__ __launch_bounds__(256)
void softmax_kernel(float* S)
{
    __shared__ float red[256];
    const long row = blockIdx.x;              // 0..8191 (batch*seq)
    float* Srow = S + row * 2048;
    u16*   Prow = (u16*)Srow;                 // P stride = 4096 bf16 elements
    const int tid = threadIdx.x;
    const float scale = 0.03125f;             // 1/sqrt(1024)

    float x[8];
    float mx = -3.402823466e38f;
#pragma unroll
    for (int i = 0; i < 8; ++i) {
        x[i] = Srow[tid + i * 256] * scale;
        mx = fmaxf(mx, x[i]);
    }
    red[tid] = mx; __syncthreads();
    for (int s = 128; s > 0; s >>= 1) {
        if (tid < s) red[tid] = fmaxf(red[tid], red[tid + s]);
        __syncthreads();
    }
    const float rowmax = red[0];
    __syncthreads();

    float sum = 0.f;
#pragma unroll
    for (int i = 0; i < 8; ++i) { x[i] = __expf(x[i] - rowmax); sum += x[i]; }
    red[tid] = sum; __syncthreads();
    for (int s = 128; s > 0; s >>= 1) {
        if (tid < s) red[tid] += red[tid + s];
        __syncthreads();
    }
    const float inv = 1.f / red[0];
#pragma unroll
    for (int i = 0; i < 8; ++i)
        Prow[tid + i * 256] = f2bf(x[i] * inv);
}

// ---------- Stage 4: att = P_b @ V_b + V_b  (NN GEMM, f32 out with residual) -
__global__ __launch_bounds__(256)
void pv_kernel(const u16* P, const u16* V, float* Out)
{
    __shared__ u32 ldsA[2][128][20];
    __shared__ u16 ldsB[2][32][136];   // 272B row stride: 16B aligned

    const long b = blockIdx.z;
    const u16* Pb = P + b * 2048L * 4096;     // bf16 P over f32 S rows, lda = 4096
    const u16* Vb = V + b * 2048L * 1024;
    float*     Ob = Out + b * 2048L * 1024;

    const int tid  = threadIdx.x;
    const int lane = tid & 31;
    const int w    = tid >> 5;
    const int wm   = (w >> 2) * 64;
    const int wn   = (w & 3) * 32;
    const int hi   = lane >> 4;
    const int ln   = lane & 15;
    const long mBase = (long)blockIdx.y * 128;
    const long nBase = (long)blockIdx.x * 128;

    const v8f vzero = {0.f,0.f,0.f,0.f,0.f,0.f,0.f,0.f};
    v8f acc[4][2];
#pragma unroll
    for (int i = 0; i < 4; ++i)
#pragma unroll
        for (int j = 0; j < 2; ++j) acc[i][j] = vzero;

    const int sm = tid >> 1, sh = (tid & 1) * 8;   // A staging
    const int bk = tid >> 3, bn = (tid & 7) * 16;  // B staging (k-row, n-chunk)

    auto stage = [&](int buf, int k0) {
        const u16* A  = Pb + (mBase + sm) * 4096L + k0 + sh * 2;
        const u16* Bv = Vb + (k0 + bk) * 1024L + nBase + bn;
#if USE_ASYNC
        async_copy_b128(A,      &ldsA[buf][sm][sh]);
        async_copy_b128(A + 8,  &ldsA[buf][sm][sh + 4]);
        async_copy_b128(Bv,     &ldsB[buf][bk][bn]);
        async_copy_b128(Bv + 8, &ldsB[buf][bk][bn + 8]);
#else
        const u32* A32 = (const u32*)A;
        const u32* B32 = (const u32*)Bv;
#pragma unroll
        for (int j = 0; j < 8; ++j) ldsA[buf][sm][sh + j] = A32[j];
        u32* dst = (u32*)(&ldsB[buf][bk][bn]);
#pragma unroll
        for (int j = 0; j < 8; ++j) dst[j] = B32[j];
        __builtin_prefetch(A32 + 16, 0, 0);
        __builtin_prefetch(B32 + 8192, 0, 0);
#endif
    };

    stage(0, 0);
#if USE_ASYNC
    __builtin_amdgcn_s_wait_asynccnt(0);
#endif
    __syncthreads();

    int buf = 0;
    for (int k0 = 0; k0 < 2048; k0 += 32, buf ^= 1) {
        const bool hasNext = (k0 + 32 < 2048);
        if (hasNext) stage(buf ^ 1, k0 + 32);

        Frag a[4], bf[2];
#pragma unroll
        for (int i = 0; i < 4; ++i) {
            const int row = wm + 16 * i + ln;
#pragma unroll
            for (int j = 0; j < 8; ++j) {
                const int k2 = hi * 8 + j;
                a[i].u[j] = ldsA[buf][row][k2];
            }
        }
#pragma unroll
        for (int i = 0; i < 2; ++i) {
            const int n = wn + 16 * i + ln;
#pragma unroll
            for (int j = 0; j < 8; ++j) {
                const int k = hi * 16 + 2 * j;
                u32 lo  = ldsB[buf][k][n];
                u32 hi2 = ldsB[buf][k + 1][n];
                bf[i].u[j] = lo | (hi2 << 16);
            }
        }
#pragma unroll
        for (int i = 0; i < 4; ++i)
#pragma unroll
            for (int j = 0; j < 2; ++j)
                acc[i][j] = __builtin_amdgcn_wmma_f32_16x16x32_bf16(
                    false, a[i].v, false, bf[j].v, (short)0, acc[i][j], false, false);

#if USE_ASYNC
        if (hasNext) __builtin_amdgcn_s_wait_asynccnt(0);
#endif
        __syncthreads();
    }

    // epilogue with residual +V
#pragma unroll
    for (int i = 0; i < 4; ++i)
#pragma unroll
        for (int jn = 0; jn < 2; ++jn) {
            const long m0 = mBase + wm + 16 * i;
            const long n  = nBase + wn + 16 * jn + ln;
#pragma unroll
            for (int r = 0; r < 8; ++r) {
                const long m = m0 + r + hi * 8;
                Ob[m * 1024L + n] = acc[i][jn][r] + bf2f(Vb[m * 1024L + n]);
            }
        }
}

// =====================================================================
// ws layout (112 MB total):
//   [0,16M)    Q bf16        [16M,32M)  K bf16      [32M,48M)  V bf16
//   [48M,112M) S f32 (64 MB) -- also hosts, transiently BEFORE score runs:
//       [48M,64M)  Xbf bf16 (16 MB)    [64M,70M)  Wbf bf16 (6 MB)
//   (Xbf/Wbf are dead once qkv_kernel finishes; score_kernel then writes S.)
// =====================================================================
extern "C" void kernel_launch(void* const* d_in, const int* in_sizes, int n_in,
                              void* d_out, int out_size, void* d_ws, size_t ws_size,
                              hipStream_t stream)
{
    const float* x  = (const float*)d_in[0];
    const float* Wq = (const float*)d_in[1];
    const float* Wk = (const float*)d_in[2];
    const float* Wv = (const float*)d_in[3];

    char* ws = (char*)d_ws;
    u16*   Q   = (u16*)(ws);
    u16*   Kd  = (u16*)(ws + (16L << 20));
    u16*   V   = (u16*)(ws + (32L << 20));
    float* S   = (float*)(ws + (48L << 20));
    u32*   Xbf = (u32*)(ws + (48L << 20));            // transient, inside S region
    u32*   Wbf = (u32*)(ws + (64L << 20));            // transient, inside S region
    float* Out = (float*)d_out;

    dim3 blk(256);
    // f32 -> bf16 bulk converts (X: 4M pairs, each W: 512K pairs)
    cvt_kernel<<<dim3(4096), blk, 0, stream>>>(x,  Xbf);
    cvt_kernel<<<dim3(512),  blk, 0, stream>>>(Wq, Wbf);
    cvt_kernel<<<dim3(512),  blk, 0, stream>>>(Wk, Wbf + 524288);
    cvt_kernel<<<dim3(512),  blk, 0, stream>>>(Wv, Wbf + 1048576);

    qkv_kernel    <<<dim3(8, 64, 3),  blk, 0, stream>>>((const u16*)Xbf, (const u16*)Wbf, Q, Kd, V);
    score_kernel  <<<dim3(16, 16, 4), blk, 0, stream>>>(Q, Kd, S);
    softmax_kernel<<<dim3(8192),      blk, 0, stream>>>(S);
    pv_kernel     <<<dim3(8, 16, 4),  blk, 0, stream>>>((const u16*)S, V, Out);
}
